// Attractor_76957224010413
// MI455X (gfx1250) — matrix operand order repaired and checked by reading, not compiled, not run on hardware
//
#include <hip/hip_runtime.h>
#include <hip/hip_bf16.h>

#define D       8192
#define TAU     16
#define SLOPE   0.01f
#define EPS     1e-12f
#define KSPLIT  8

typedef __bf16 bf16;
typedef __attribute__((ext_vector_type(16))) __bf16 v16bf;
typedef __attribute__((ext_vector_type(8)))  __bf16 v8bf;
typedef __attribute__((ext_vector_type(8)))  float  v8f;

// ---------------------------------------------------------------------------
// 1) One-time: Mt[n][k] = bf16(M[k][n]) via LDS-tiled transpose.
//    Row-major-by-n Mt makes WMMA B-fragment loads contiguous in K, and at
//    bf16 the whole matrix (128 MB) fits in the 192 MB L2 for all 16 steps.
// ---------------------------------------------------------------------------
__global__ void conv_transpose_kernel(const float* __restrict__ M,
                                      bf16* __restrict__ Mt) {
    __shared__ float tile[32][33];
    const int n0 = blockIdx.x * 32;
    const int k0 = blockIdx.y * 32;
    const int tx = threadIdx.x;   // 0..31
    const int ty = threadIdx.y;   // 0..7
#pragma unroll
    for (int j = 0; j < 4; ++j)
        tile[ty + j * 8][tx] = M[(size_t)(k0 + ty + j * 8) * D + n0 + tx];
    __syncthreads();
#pragma unroll
    for (int j = 0; j < 4; ++j)
        Mt[(size_t)(n0 + ty + j * 8) * D + k0 + tx] = (bf16)tile[tx][ty + j * 8];
}

// ---------------------------------------------------------------------------
// 2) Init: A rows 0..7 = bf16(x), rows 8..15 = 0 (WMMA padding); hs = 0.
// ---------------------------------------------------------------------------
__global__ void init_kernel(const float* __restrict__ x,
                            float* __restrict__ hs,
                            bf16* __restrict__ A) {
    const int idx = blockIdx.x * blockDim.x + threadIdx.x;  // 0 .. 16*D-1
    if (idx < 8 * D) {
        A[idx]  = (bf16)x[idx];
        hs[idx] = 0.0f;
    } else {
        A[idx] = (bf16)0.0f;
    }
}

// ---------------------------------------------------------------------------
// 3) GEMM step: one wave computes a 16x16 f32 tile of (A @ Mt^T) over a
//    1024-wide K slice using v_wmma_f32_16x16x32_bf16, streaming Mt from L2.
//    grid = (512 column tiles, KSPLIT), block = 32 (one wave).
//    Unroll-by-4 keeps 16 b128 loads in flight per wave to cover L2 latency.
// ---------------------------------------------------------------------------
__global__ void __launch_bounds__(32)
gemm_kernel(const bf16* __restrict__ A,
            const bf16* __restrict__ Mt,
            float* __restrict__ pre_part) {
    const int lane = threadIdx.x;        // 0..31
    const int n0   = blockIdx.x * 16;    // output column tile
    const int ks   = blockIdx.y;         // K split 0..KSPLIT-1
    const int idx16 = lane & 15;
    const int hi    = lane >> 4;

    // A fragment: row = idx16; K pieces [koff,koff+8) and [koff+16,koff+24)
    const bf16* Arow = A + (size_t)idx16 * D + hi * 8;
    // B fragment: column n0+idx16 of M == row of Mt; K pieces [koff,koff+8) and [koff+16,koff+24)
    const bf16* Brow = Mt + (size_t)(n0 + idx16) * D + hi * 8;

    v8f acc = {};
    const int kbeg = ks * (D / KSPLIT);
    const int kend = kbeg + (D / KSPLIT);
#pragma unroll 4
    for (int k = kbeg; k < kend; k += 32) {
        v8bf alo = *(const v8bf*)(Arow + k);
        v8bf ahi = *(const v8bf*)(Arow + k + 16);
        v8bf b0  = *(const v8bf*)(Brow + k);
        v8bf b1  = *(const v8bf*)(Brow + k + 16);
        v16bf a, b;
#pragma unroll
        for (int i = 0; i < 8; ++i) {
            a[i] = alo[i]; a[i + 8] = ahi[i];
            b[i] = b0[i];  b[i + 8] = b1[i];
        }
        acc = __builtin_amdgcn_wmma_f32_16x16x32_bf16(
            /*neg_a=*/false, a, /*neg_b=*/false, b,
            /*c_mod=*/(short)0, acc, /*reuse_a=*/false, /*reuse_b=*/false);
    }

    // C/D layout: lane(0..15) holds column n0+lane, VGPR r -> row r.
    // Only rows 0..7 are real (rows 8..15 are zero padding on lanes 16..31).
    if (lane < 16) {
        float* dst = pre_part + (size_t)ks * 8 * D + n0 + lane;
#pragma unroll
        for (int r = 0; r < 8; ++r)
            dst[(size_t)r * D] = acc[r];
    }
}

// ---------------------------------------------------------------------------
// 4) Normalize step: pre = 0.5*hs + sum(partials); act = leaky_relu(pre);
//    hs_new = act / max(||act||_2, eps). One block per row; act staged in LDS.
//    Writes f32 hs (d_out) and bf16 A for the next step's WMMA A-fragment.
// ---------------------------------------------------------------------------
__global__ void __launch_bounds__(512)
norm_kernel(const float* __restrict__ pre_part,
            float* __restrict__ hs,
            bf16* __restrict__ A) {
    __shared__ float act[D];       // 32 KB of the 320 KB WGP LDS
    __shared__ float red[512];
    const int r   = blockIdx.x;    // 0..7
    const int tid = threadIdx.x;   // 0..511

    float s = 0.0f;
#pragma unroll
    for (int i = 0; i < D / 512; ++i) {
        const int c = tid + i * 512;
        float pre = 0.5f * hs[(size_t)r * D + c];
#pragma unroll
        for (int p = 0; p < KSPLIT; ++p)
            pre += pre_part[((size_t)p * 8 + r) * D + c];
        const float a = (pre >= 0.0f) ? pre : SLOPE * pre;
        act[c] = a;
        s += a * a;
    }
    red[tid] = s;
    __syncthreads();
    for (int off = 256; off > 0; off >>= 1) {
        if (tid < off) red[tid] += red[tid + off];
        __syncthreads();
    }
    const float inv = 1.0f / fmaxf(sqrtf(red[0]), EPS);
#pragma unroll
    for (int i = 0; i < D / 512; ++i) {
        const int c = tid + i * 512;
        const float v = act[c] * inv;
        hs[(size_t)r * D + c] = v;
        A[(size_t)r * D + c]  = (bf16)v;
    }
}

// ---------------------------------------------------------------------------
extern "C" void kernel_launch(void* const* d_in, const int* in_sizes, int n_in,
                              void* d_out, int out_size, void* d_ws, size_t ws_size,
                              hipStream_t stream) {
    const float* x = (const float*)d_in[0];   // (8, 8192)
    const float* M = (const float*)d_in[1];   // (8192, 8192)
    // d_in[2] (hs) is known zeros; handled by init_kernel.

    float* hs = (float*)d_out;                // (8, 8192) — final answer lands here

    char* ws = (char*)d_ws;
    bf16* Mt = (bf16*)ws;                                        // 128 MB
    bf16* A  = (bf16*)(ws + (size_t)D * D * sizeof(bf16));       // 256 KB (16 x D)
    float* pre_part =
        (float*)(ws + (size_t)D * D * sizeof(bf16) + (size_t)16 * D * sizeof(bf16)); // 2 MB

    // One-time per launch: bf16 transpose of M (idempotent, graph-safe).
    conv_transpose_kernel<<<dim3(D / 32, D / 32), dim3(32, 8), 0, stream>>>(M, Mt);
    init_kernel<<<(16 * D) / 256, 256, 0, stream>>>(x, hs, A);

    for (int t = 0; t < TAU; ++t) {
        gemm_kernel<<<dim3(D / 16, KSPLIT), 32, 0, stream>>>(A, Mt, pre_part);
        norm_kernel<<<8, 512, 0, stream>>>(pre_part, hs, A);
    }
}
